// CompletionShadowNet_73023033967248
// MI455X (gfx1250) — matrix operand order repaired and checked by reading, not compiled, not run on hardware
//
#include <hip/hip_runtime.h>

typedef __attribute__((ext_vector_type(16))) __bf16          v16bf;
typedef __attribute__((ext_vector_type(16))) unsigned short  v16u;
typedef __attribute__((ext_vector_type(8)))  unsigned short  u16x8;
typedef __attribute__((ext_vector_type(8)))  float           v8f;

__device__ __forceinline__ unsigned short f2bf(float f) {
  unsigned u = __builtin_bit_cast(unsigned, f);
  u = u + 0x7FFFu + ((u >> 16) & 1u);          // round-to-nearest-even
  return (unsigned short)(u >> 16);
}
__device__ __forceinline__ float bf2f(unsigned short s) {
  unsigned u = ((unsigned)s) << 16;
  return __builtin_bit_cast(float, u);
}
#define CAT16(lo, hi) __builtin_shufflevector(lo, hi, 0, 1, 2, 3, 4, 5, 6, 7, 8, 9, 10, 11, 12, 13, 14, 15)

// ---------------------------------------------------------------------------
// conv3 (k=3, s=1, SAME) implicit GEMM on WMMA bf16, fused mask + BN stats.
// A: 16 voxels x 32 k (tap*CIN+c); B: 32 k x 16 couts (weights pre-transposed
// to [COUT][KPAD] so each lane's B fragment is 2 contiguous b128 loads).
// Grid is exact (nvox % 64 == 0): no bounds guards anywhere.
// ---------------------------------------------------------------------------
template <int CIN, int COUT, int D, int H, int W>
__global__ __launch_bounds__(128) void conv3_wmma(
    const unsigned short* __restrict__ act, const unsigned char* __restrict__ mask,
    const unsigned short* __restrict__ wts, float* __restrict__ y,
    float* __restrict__ stats) {
  constexpr int KTOT = 27 * CIN;
  constexpr int KPAD = (KTOT + 31) & ~31;
  static_assert((2 * D * H * W) % 64 == 0, "tile-exact");
  const int lane  = threadIdx.x & 31;
  const int vbase = (blockIdx.x * 4 + (threadIdx.x >> 5)) * 16;
  const int n0    = blockIdx.y * 16;

  __shared__ float ssum[16], ssq[16];
  if (threadIdx.x < 16) { ssum[threadIdx.x] = 0.f; ssq[threadIdx.x] = 0.f; }
  __syncthreads();

  const int vi = vbase + (lane & 15);
  int t = vi;
  const int w = t % W; t /= W;
  const int h = t % H; t /= H;
  const int d = t % D; const int b = t / D;
  const int kb = (lane >> 4) * 8;                         // A k-base (runs of 8)
  const unsigned short* wrow = wts + (size_t)(n0 + (lane & 15)) * KPAD + (lane >> 4) * 8;

  v8f acc = {};
  for (int kc = 0; kc < KPAD; kc += 32) {
    v16u au;
    if constexpr (CIN >= 8) {
      u16x8 lo = {}, hi = {};
#pragma unroll
      for (int half = 0; half < 2; ++half) {
        const int kg = kc + kb + 16 * half;               // run of 8 consecutive k
        if (kg < KTOT) {
          const int tap = kg / CIN, c0 = kg - tap * CIN;  // run stays in one tap
          const int dz = tap / 9 - 1, dy = (tap / 3) % 3 - 1, dx = tap % 3 - 1;
          const int zz = d + dz, yy = h + dy, xx = w + dx;
          if ((unsigned)zz < (unsigned)D && (unsigned)yy < (unsigned)H &&
              (unsigned)xx < (unsigned)W) {
            const u16x8 v =
                *(const u16x8*)(act + ((((size_t)b * D + zz) * H + yy) * (size_t)W + xx) * CIN + c0);
            if (half) hi = v; else lo = v;
          }
        }
      }
      au = CAT16(lo, hi);
    } else {                                              // CIN==1 (e1): scalar gather
#pragma unroll
      for (int j = 0; j < 16; ++j) {
        const int kg = kc + kb + (j >> 3) * 16 + (j & 7);
        float av = 0.f;
        if (kg < KTOT) {
          const int dz = kg / 9 - 1, dy = (kg / 3) % 3 - 1, dx = kg % 3 - 1;
          const int zz = d + dz, yy = h + dy, xx = w + dx;
          if ((unsigned)zz < (unsigned)D && (unsigned)yy < (unsigned)H &&
              (unsigned)xx < (unsigned)W)
            av = bf2f(act[(((size_t)b * D + zz) * H + yy) * (size_t)W + xx]);
        }
        au[j] = f2bf(av);
      }
    }
    const u16x8 b0 = *(const u16x8*)(wrow + kc);
    const u16x8 b1 = *(const u16x8*)(wrow + kc + 16);
    acc = __builtin_amdgcn_wmma_f32_16x16x32_bf16(
        false, __builtin_bit_cast(v16bf, au),
        false, __builtin_bit_cast(v16bf, CAT16(b0, b1)), (short)0, acc, false, false);
  }

  // epilogue: 8 consecutive voxels per lane, one 8-byte mask load, no branches
  const int ch  = n0 + (lane & 15);
  const int vvb = vbase + 8 * (lane >> 4);
  const unsigned long long mk8 = *(const unsigned long long*)(mask + vvb);
  float* yp = y + (size_t)vvb * COUT + ch;
  float ls = 0.f, lq = 0.f;
#pragma unroll
  for (int r = 0; r < 8; ++r) {
    const float val = ((mk8 >> (8 * r)) & 0xFFull) ? acc[r] : 0.f;
    yp[(size_t)r * COUT] = val;
    ls += val; lq += val * val;
  }
  atomicAdd(&ssum[lane & 15], ls);
  atomicAdd(&ssq[lane & 15], lq);
  __syncthreads();
  if (threadIdx.x < 16) {
    atomicAdd(&stats[n0 + threadIdx.x], ssum[threadIdx.x]);
    atomicAdd(&stats[COUT + n0 + threadIdx.x], ssq[threadIdx.x]);
  }
}

// ---------------------------------------------------------------------------
// down2 (k=2, s=2): K = 8*CIN (multiple of 32). Dims are COARSE dims.
// ---------------------------------------------------------------------------
template <int CIN, int COUT, int D, int H, int W>
__global__ __launch_bounds__(128) void down2_wmma(
    const unsigned short* __restrict__ act, const unsigned char* __restrict__ mask,
    const unsigned short* __restrict__ wts, float* __restrict__ y,
    float* __restrict__ stats) {
  constexpr int KTOT = 8 * CIN;
  const int lane  = threadIdx.x & 31;
  const int vbase = (blockIdx.x * 4 + (threadIdx.x >> 5)) * 16;
  const int n0    = blockIdx.y * 16;

  __shared__ float ssum[16], ssq[16];
  if (threadIdx.x < 16) { ssum[threadIdx.x] = 0.f; ssq[threadIdx.x] = 0.f; }
  __syncthreads();

  const int vi = vbase + (lane & 15);
  int t = vi;
  const int w = t % W; t /= W;
  const int h = t % H; t /= H;
  const int d = t % D; const int b = t / D;
  const int kb = (lane >> 4) * 8;
  constexpr int D2 = 2 * D, H2 = 2 * H, W2 = 2 * W;
  const unsigned short* wrow = wts + (size_t)(n0 + (lane & 15)) * KTOT + (lane >> 4) * 8;

  v8f acc = {};
  for (int kc = 0; kc < KTOT; kc += 32) {
    u16x8 part[2];
#pragma unroll
    for (int half = 0; half < 2; ++half) {
      const int kg  = kc + kb + 16 * half;
      const int tap = kg / CIN, c0 = kg - (kg / CIN) * CIN;
      const int xb = (tap >> 2) & 1, yb = (tap >> 1) & 1, zb = tap & 1;
      part[half] = *(const u16x8*)(act +
          ((((size_t)b * D2 + 2 * d + xb) * H2 + 2 * h + yb) * (size_t)W2 + 2 * w + zb) * CIN + c0);
    }
    const u16x8 b0 = *(const u16x8*)(wrow + kc);
    const u16x8 b1 = *(const u16x8*)(wrow + kc + 16);
    acc = __builtin_amdgcn_wmma_f32_16x16x32_bf16(
        false, __builtin_bit_cast(v16bf, CAT16(part[0], part[1])),
        false, __builtin_bit_cast(v16bf, CAT16(b0, b1)), (short)0, acc, false, false);
  }

  const int ch  = n0 + (lane & 15);
  const int vvb = vbase + 8 * (lane >> 4);
  const unsigned long long mk8 = *(const unsigned long long*)(mask + vvb);
  float* yp = y + (size_t)vvb * COUT + ch;
  float ls = 0.f, lq = 0.f;
#pragma unroll
  for (int r = 0; r < 8; ++r) {
    const float val = ((mk8 >> (8 * r)) & 0xFFull) ? acc[r] : 0.f;
    yp[(size_t)r * COUT] = val;
    ls += val; lq += val * val;
  }
  atomicAdd(&ssum[lane & 15], ls);
  atomicAdd(&ssq[lane & 15], lq);
  __syncthreads();
  if (threadIdx.x < 16) {
    atomicAdd(&stats[n0 + threadIdx.x], ssum[threadIdx.x]);
    atomicAdd(&stats[COUT + n0 + threadIdx.x], ssq[threadIdx.x]);
  }
}

// ---------------------------------------------------------------------------
// up2 (generative transpose k=2 s=2): 8 GEMMs (one per child tap) sharing the
// A fragment. Dims are COARSE dims; y/mask are fine-level.
// ---------------------------------------------------------------------------
template <int CIN, int COUT, int D, int H, int W>
__global__ __launch_bounds__(64) void up2_wmma(
    const unsigned short* __restrict__ act, const unsigned char* __restrict__ maskf,
    const unsigned short* __restrict__ wts, float* __restrict__ y,
    float* __restrict__ stats) {
  const int lane  = threadIdx.x & 31;
  const int vbase = (blockIdx.x * 2 + (threadIdx.x >> 5)) * 16;
  const int n0    = blockIdx.y * 16;
  constexpr int D2 = 2 * D, H2 = 2 * H, W2 = 2 * W;
  constexpr int KTOT = 8 * CIN;

  __shared__ float ssum[16], ssq[16];
  if (threadIdx.x < 16) { ssum[threadIdx.x] = 0.f; ssq[threadIdx.x] = 0.f; }
  __syncthreads();

  const int vi = vbase + (lane & 15);
  const int kb = (lane >> 4) * 8;
  const unsigned short* wrow = wts + (size_t)(n0 + (lane & 15)) * KTOT + (lane >> 4) * 8;

  v8f zero = {};
  v8f acc[8];
#pragma unroll
  for (int tp = 0; tp < 8; ++tp) acc[tp] = zero;

  for (int kc = 0; kc < CIN; kc += 32) {
    const u16x8 a0 = *(const u16x8*)(act + (size_t)vi * CIN + kc + kb);
    const u16x8 a1 = *(const u16x8*)(act + (size_t)vi * CIN + kc + kb + 16);
    const v16bf af = __builtin_bit_cast(v16bf, CAT16(a0, a1));
#pragma unroll
    for (int tp = 0; tp < 8; ++tp) {
      const u16x8 b0 = *(const u16x8*)(wrow + tp * CIN + kc);
      const u16x8 b1 = *(const u16x8*)(wrow + tp * CIN + kc + 16);
      acc[tp] = __builtin_amdgcn_wmma_f32_16x16x32_bf16(
          false, af, false, __builtin_bit_cast(v16bf, CAT16(b0, b1)), (short)0,
          acc[tp], false, false);
    }
  }

  float ls = 0.f, lq = 0.f;
  const int ch = n0 + (lane & 15);
#pragma unroll
  for (int r = 0; r < 8; ++r) {
    const int vv = vbase + 8 * (lane >> 4) + r;
    int t2 = vv;
    const int ww = t2 % W; t2 /= W;
    const int hh = t2 % H; t2 /= H;
    const int dd = t2 % D; const int bb = t2 / D;
#pragma unroll
    for (int tp = 0; tp < 8; ++tp) {
      const int xb = (tp >> 2) & 1, yb = (tp >> 1) & 1, zb = tp & 1;
      const size_t cf = (((size_t)bb * D2 + 2 * dd + xb) * H2 + 2 * hh + yb) * (size_t)W2 +
                        2 * ww + zb;
      const float val = maskf[cf] ? acc[tp][r] : 0.f;
      y[cf * COUT + ch] = val;
      ls += val; lq += val * val;
    }
  }
  atomicAdd(&ssum[lane & 15], ls);
  atomicAdd(&ssq[lane & 15], lq);
  __syncthreads();
  if (threadIdx.x < 16) {
    atomicAdd(&stats[n0 + threadIdx.x], ssum[threadIdx.x]);
    atomicAdd(&stats[COUT + n0 + threadIdx.x], ssq[threadIdx.x]);
  }
}

// ---------------------------------------------------------------------------
// BN (masked stats) + ELU + mask, f32 -> bf16
// ---------------------------------------------------------------------------
template <int COUT>
__global__ void bn_elu(const float* __restrict__ y, const float* __restrict__ stats,
                       const int* __restrict__ cnt, const float* __restrict__ g,
                       const float* __restrict__ bb, const unsigned char* __restrict__ mask,
                       unsigned short* __restrict__ outb, int nvox) {
  const long long i = (long long)blockIdx.x * 256 + threadIdx.x;
  if (i >= (long long)nvox * COUT) return;
  const int ch = (int)(i % COUT);
  const int vi = (int)(i / COUT);
  const float c   = fmaxf((float)*cnt, 1.f);
  const float mu  = stats[ch] / c;
  const float var = fmaxf(stats[COUT + ch] / c - mu * mu, 0.f);
  float val = (y[i] - mu) * rsqrtf(var + 1e-5f) * g[ch] + bb[ch];
  val = val > 0.f ? val : (__expf(val) - 1.f);
  if (!mask[vi]) val = 0.f;
  outb[i] = f2bf(val);
}

// ---------------------------------------------------------------------------
// classification head + pruning
// ---------------------------------------------------------------------------
template <int CIN>
__global__ void cls_prune(unsigned short* __restrict__ act, const unsigned char* __restrict__ mask,
                          const float* __restrict__ wc, const float* __restrict__ bc,
                          float* __restrict__ cout_, unsigned char* __restrict__ kbuf, int nvox) {
  const int vi = blockIdx.x * 256 + threadIdx.x;
  if (vi >= nvox) return;
  float c = bc[0];
#pragma unroll
  for (int ch = 0; ch < CIN; ++ch) c += bf2f(act[(size_t)vi * CIN + ch]) * wc[ch];
  const bool mk = mask[vi] != 0;
  if (!mk) c = 0.f;
  cout_[vi] = c;
  const bool k = (c > 0.f) && mk;
  kbuf[vi] = k ? 1 : 0;
  if (!k)
#pragma unroll
    for (int ch = 0; ch < CIN; ++ch) act[(size_t)vi * CIN + ch] = 0;
}

// ---------------------------------------------------------------------------
// elementwise / mask utility kernels
// ---------------------------------------------------------------------------
__global__ void cvt_w(const float* __restrict__ s, unsigned short* __restrict__ d,
                      int K, int KPAD, int COUT) {           // -> transposed [COUT][KPAD]
  const int i = blockIdx.x * 256 + threadIdx.x;
  if (i >= KPAD * COUT) return;
  const int k = i % KPAD, n = i / KPAD;
  d[i] = (k < K) ? f2bf(s[(size_t)k * COUT + n]) : (unsigned short)0;
}
__global__ void cvt_x(const float* __restrict__ s, unsigned short* __restrict__ d, int n) {
  const int i = blockIdx.x * 256 + threadIdx.x;
  if (i < n) d[i] = f2bf(s[i]);
}
__global__ void zero_f32(float* p, int n) {
  const int i = blockIdx.x * 256 + threadIdx.x;
  if (i < n) p[i] = 0.f;
}
__global__ void zero_i32(int* p, int n) {
  const int i = blockIdx.x * 256 + threadIdx.x;
  if (i < n) p[i] = 0;
}
__global__ void pool_any2(const unsigned char* __restrict__ in, unsigned char* __restrict__ outm,
                          int Bn, int D, int H, int W) {   // coarse dims
  const int vi = blockIdx.x * 256 + threadIdx.x;
  const int n  = Bn * D * H * W;
  if (vi >= n) return;
  int t = vi;
  const int w = t % W; t /= W;
  const int h = t % H; t /= H;
  const int d = t % D; const int b = t / D;
  const int D2 = 2 * D, H2 = 2 * H, W2 = 2 * W;
  unsigned m = 0;
  for (int z = 0; z < 2; ++z)
    for (int yy = 0; yy < 2; ++yy)
      for (int x = 0; x < 2; ++x)
        m |= in[(((size_t)b * D2 + 2 * d + z) * H2 + 2 * h + yy) * (size_t)W2 + 2 * w + x];
  outm[vi] = m ? 1 : 0;
}
__global__ void up_mask_k(const unsigned char* __restrict__ in, unsigned char* __restrict__ outm,
                          int Bn, int D, int H, int W) {   // coarse dims; out is fine
  const int D2 = 2 * D, H2 = 2 * H, W2 = 2 * W;
  const long long n = (long long)Bn * D2 * H2 * W2;
  const long long i = (long long)blockIdx.x * 256 + threadIdx.x;
  if (i >= n) return;
  long long t = i;
  const int w = (int)(t % W2); t /= W2;
  const int h = (int)(t % H2); t /= H2;
  const int d = (int)(t % D2); const int b = (int)(t / D2);
  outm[i] = in[(((size_t)b * D + (d >> 1)) * H + (h >> 1)) * (size_t)W + (w >> 1)];
}
__global__ void count_mask(const unsigned char* __restrict__ m, int n, int* __restrict__ cnt) {
  __shared__ int s;
  if (threadIdx.x == 0) s = 0;
  __syncthreads();
  const int i = blockIdx.x * 256 + threadIdx.x;
  const int v = (i < n && m[i]) ? 1 : 0;
  atomicAdd(&s, v);
  __syncthreads();
  if (threadIdx.x == 0 && s) atomicAdd(cnt, s);
}
__global__ void t_mask(const unsigned char* __restrict__ md, const unsigned char* __restrict__ tgt,
                       float* __restrict__ outf, int Bn, int D, int H, int W, int f) {
  const int vi = blockIdx.x * 256 + threadIdx.x;
  const int n  = Bn * D * H * W;
  if (vi >= n) return;
  int t = vi;
  const int w = t % W; t /= W;
  const int h = t % H; t /= H;
  const int d = t % D; const int b = t / D;
  const int Df = D * f, Hf = H * f, Wf = W * f;
  unsigned any = 0;
  for (int z = 0; z < f; ++z)
    for (int yy = 0; yy < f; ++yy)
      for (int x = 0; x < f; ++x)
        any |= tgt[(((size_t)b * Df + d * f + z) * Hf + h * f + yy) * (size_t)Wf + w * f + x];
  outf[vi] = (md[vi] && any) ? 1.f : 0.f;
}
__global__ void write_bf_to_f32(const unsigned short* __restrict__ a, float* __restrict__ o,
                                long long n) {
  const long long i = (long long)blockIdx.x * 256 + threadIdx.x;
  if (i < n) o[i] = bf2f(a[i]);
}
__global__ void write_mask_f32(const unsigned char* __restrict__ m, float* __restrict__ o, int n) {
  const int i = blockIdx.x * 256 + threadIdx.x;
  if (i < n) o[i] = m[i] ? 1.f : 0.f;
}

// ---------------------------------------------------------------------------
// host orchestration
// ---------------------------------------------------------------------------
static inline dim3 eb(long long n) { return dim3((unsigned)((n + 255) / 256)); }
static inline dim3 cg(int nvox, int cout, int waves) {
  const int tiles = nvox / 16;                 // exact at every level
  return dim3((unsigned)(tiles / waves), (unsigned)(cout / 16));
}

extern "C" void kernel_launch(void* const* d_in, const int* in_sizes, int n_in,
                              void* d_out, int out_size, void* d_ws, size_t ws_size,
                              hipStream_t stream) {
  (void)in_sizes; (void)n_in; (void)out_size; (void)ws_size;
  const float* Pf[48];
  for (int i = 0; i < 48; ++i) Pf[i] = (const float*)d_in[i];
  const float*         x   = (const float*)d_in[45];
  const unsigned char* m1  = (const unsigned char*)d_in[46];
  const unsigned char* tgt = (const unsigned char*)d_in[47];
  float* out = (float*)d_out;
  char*  ws  = (char*)d_ws;

  constexpr int Bn  = 2;
  constexpr int NV1 = 2 * 96 * 96 * 96, NV2 = 2 * 48 * 48 * 48;
  constexpr int NV4 = 2 * 24 * 24 * 24, NV8 = 2 * 12 * 12 * 12;

  size_t off = 0;
  auto alloc = [&](size_t bytes) -> char* {
    char* p = ws + off;
    off = (off + bytes + 255) & ~(size_t)255;
    return p;
  };
  unsigned short* wE1   = (unsigned short*)alloc((size_t)32 * 16 * 2);
  unsigned short* wE12a = (unsigned short*)alloc((size_t)128 * 32 * 2);
  unsigned short* wE12b = (unsigned short*)alloc((size_t)864 * 32 * 2);
  unsigned short* wE24a = (unsigned short*)alloc((size_t)256 * 64 * 2);
  unsigned short* wE24b = (unsigned short*)alloc((size_t)1728 * 64 * 2);
  unsigned short* wE48a = (unsigned short*)alloc((size_t)512 * 128 * 2);
  unsigned short* wE48b = (unsigned short*)alloc((size_t)3456 * 128 * 2);
  unsigned short* wD84a = (unsigned short*)alloc((size_t)1024 * 64 * 2);
  unsigned short* wD84b = (unsigned short*)alloc((size_t)1728 * 64 * 2);
  unsigned short* wD42a = (unsigned short*)alloc((size_t)512 * 32 * 2);
  unsigned short* wD42b = (unsigned short*)alloc((size_t)864 * 32 * 2);
  unsigned short* wD21a = (unsigned short*)alloc((size_t)256 * 16 * 2);
  unsigned short* wD21b = (unsigned short*)alloc((size_t)448 * 16 * 2);
  unsigned short* xbf   = (unsigned short*)alloc((size_t)NV1 * 2);
  unsigned short* actA  = (unsigned short*)alloc((size_t)NV1 * 16 * 2);
  unsigned short* actB  = (unsigned short*)alloc((size_t)NV1 * 16 * 2);
  float*          ybuf  = (float*)alloc((size_t)NV1 * 16 * 4);
  float*          st    = (float*)alloc(256 * 4);
  int*            cnts  = (int*)alloc(8 * 4);
  unsigned char*  m2    = (unsigned char*)alloc(NV2);
  unsigned char*  m4    = (unsigned char*)alloc(NV4);
  unsigned char*  m8    = (unsigned char*)alloc(NV8);
  unsigned char*  md4   = (unsigned char*)alloc(NV4);
  unsigned char*  k4    = (unsigned char*)alloc(NV4);
  unsigned char*  md2   = (unsigned char*)alloc(NV2);
  unsigned char*  k2    = (unsigned char*)alloc(NV2);
  unsigned char*  md1   = (unsigned char*)alloc(NV1);
  unsigned char*  k1    = (unsigned char*)alloc(NV1);

  const size_t OFF_C4 = 0, OFF_C2 = OFF_C4 + NV4, OFF_C1 = OFF_C2 + NV2;
  const size_t OFF_T4 = OFF_C1 + NV1, OFF_T2 = OFF_T4 + NV4, OFF_T1 = OFF_T2 + NV2;
  const size_t OFF_D1 = OFF_T1 + NV1, OFF_K1 = OFF_D1 + (size_t)NV1 * 16;

  // ---- weight conversion to transposed padded bf16 [COUT][KPAD] ----
  cvt_w<<<eb(32 * 16), 256, 0, stream>>>(Pf[0], wE1, 27, 32, 16);
  cvt_w<<<eb(128 * 32), 256, 0, stream>>>(Pf[3], wE12a, 128, 128, 32);
  cvt_w<<<eb(864 * 32), 256, 0, stream>>>(Pf[6], wE12b, 864, 864, 32);
  cvt_w<<<eb(256 * 64), 256, 0, stream>>>(Pf[9], wE24a, 256, 256, 64);
  cvt_w<<<eb(1728 * 64), 256, 0, stream>>>(Pf[12], wE24b, 1728, 1728, 64);
  cvt_w<<<eb(512 * 128), 256, 0, stream>>>(Pf[15], wE48a, 512, 512, 128);
  cvt_w<<<eb(3456 * 128), 256, 0, stream>>>(Pf[18], wE48b, 3456, 3456, 128);
  cvt_w<<<eb(1024 * 64), 256, 0, stream>>>(Pf[21], wD84a, 1024, 1024, 64);
  cvt_w<<<eb(1728 * 64), 256, 0, stream>>>(Pf[24], wD84b, 1728, 1728, 64);
  cvt_w<<<eb(512 * 32), 256, 0, stream>>>(Pf[27], wD42a, 512, 512, 32);
  cvt_w<<<eb(864 * 32), 256, 0, stream>>>(Pf[30], wD42b, 864, 864, 32);
  cvt_w<<<eb(256 * 16), 256, 0, stream>>>(Pf[33], wD21a, 256, 256, 16);
  cvt_w<<<eb(448 * 16), 256, 0, stream>>>(Pf[36], wD21b, 432, 448, 16);
  cvt_x<<<eb(NV1), 256, 0, stream>>>(x, xbf, NV1);

  // ---- masks + counts ----
  pool_any2<<<eb(NV2), 256, 0, stream>>>(m1, m2, Bn, 48, 48, 48);
  pool_any2<<<eb(NV4), 256, 0, stream>>>(m2, m4, Bn, 24, 24, 24);
  pool_any2<<<eb(NV8), 256, 0, stream>>>(m4, m8, Bn, 12, 12, 12);
  up_mask_k<<<eb(NV4), 256, 0, stream>>>(m8, md4, Bn, 12, 12, 12);
  zero_i32<<<1, 256, 0, stream>>>(cnts, 8);
  count_mask<<<eb(NV1), 256, 0, stream>>>(m1, NV1, cnts + 0);
  count_mask<<<eb(NV2), 256, 0, stream>>>(m2, NV2, cnts + 1);
  count_mask<<<eb(NV4), 256, 0, stream>>>(m4, NV4, cnts + 2);
  count_mask<<<eb(NV8), 256, 0, stream>>>(m8, NV8, cnts + 3);
  count_mask<<<eb(NV4), 256, 0, stream>>>(md4, NV4, cnts + 4);

  // ---- encoder ----
  zero_f32<<<1, 256, 0, stream>>>(st, 256);
  conv3_wmma<1, 16, 96, 96, 96><<<cg(NV1, 16, 4), 128, 0, stream>>>(xbf, m1, wE1, ybuf, st);
  bn_elu<16><<<eb((long long)NV1 * 16), 256, 0, stream>>>(ybuf, st, cnts + 0, Pf[1], Pf[2], m1, actA, NV1);

  zero_f32<<<1, 256, 0, stream>>>(st, 256);
  down2_wmma<16, 32, 48, 48, 48><<<cg(NV2, 32, 4), 128, 0, stream>>>(actA, m2, wE12a, ybuf, st);
  bn_elu<32><<<eb((long long)NV2 * 32), 256, 0, stream>>>(ybuf, st, cnts + 1, Pf[4], Pf[5], m2, actB, NV2);
  zero_f32<<<1, 256, 0, stream>>>(st, 256);
  conv3_wmma<32, 32, 48, 48, 48><<<cg(NV2, 32, 4), 128, 0, stream>>>(actB, m2, wE12b, ybuf, st);
  bn_elu<32><<<eb((long long)NV2 * 32), 256, 0, stream>>>(ybuf, st, cnts + 1, Pf[7], Pf[8], m2, actA, NV2);

  zero_f32<<<1, 256, 0, stream>>>(st, 256);
  down2_wmma<32, 64, 24, 24, 24><<<cg(NV4, 64, 4), 128, 0, stream>>>(actA, m4, wE24a, ybuf, st);
  bn_elu<64><<<eb((long long)NV4 * 64), 256, 0, stream>>>(ybuf, st, cnts + 2, Pf[10], Pf[11], m4, actB, NV4);
  zero_f32<<<1, 256, 0, stream>>>(st, 256);
  conv3_wmma<64, 64, 24, 24, 24><<<cg(NV4, 64, 4), 128, 0, stream>>>(actB, m4, wE24b, ybuf, st);
  bn_elu<64><<<eb((long long)NV4 * 64), 256, 0, stream>>>(ybuf, st, cnts + 2, Pf[13], Pf[14], m4, actA, NV4);

  zero_f32<<<1, 256, 0, stream>>>(st, 256);
  down2_wmma<64, 128, 12, 12, 12><<<cg(NV8, 128, 4), 128, 0, stream>>>(actA, m8, wE48a, ybuf, st);
  bn_elu<128><<<eb((long long)NV8 * 128), 256, 0, stream>>>(ybuf, st, cnts + 3, Pf[16], Pf[17], m8, actB, NV8);
  zero_f32<<<1, 256, 0, stream>>>(st, 256);
  conv3_wmma<128, 128, 12, 12, 12><<<cg(NV8, 128, 4), 128, 0, stream>>>(actB, m8, wE48b, ybuf, st);
  bn_elu<128><<<eb((long long)NV8 * 128), 256, 0, stream>>>(ybuf, st, cnts + 3, Pf[19], Pf[20], m8, actA, NV8);

  // ---- decoder level 8 -> 4 ----
  zero_f32<<<1, 256, 0, stream>>>(st, 256);
  up2_wmma<128, 64, 12, 12, 12><<<cg(NV8, 64, 2), 64, 0, stream>>>(actA, md4, wD84a, ybuf, st);
  bn_elu<64><<<eb((long long)NV4 * 64), 256, 0, stream>>>(ybuf, st, cnts + 4, Pf[22], Pf[23], md4, actB, NV4);
  zero_f32<<<1, 256, 0, stream>>>(st, 256);
  conv3_wmma<64, 64, 24, 24, 24><<<cg(NV4, 64, 4), 128, 0, stream>>>(actB, md4, wD84b, ybuf, st);
  bn_elu<64><<<eb((long long)NV4 * 64), 256, 0, stream>>>(ybuf, st, cnts + 4, Pf[25], Pf[26], md4, actA, NV4);
  cls_prune<64><<<eb(NV4), 256, 0, stream>>>(actA, md4, Pf[39], Pf[40], out + OFF_C4, k4, NV4);
  t_mask<<<eb(NV4), 256, 0, stream>>>(md4, tgt, out + OFF_T4, Bn, 24, 24, 24, 4);
  up_mask_k<<<eb(NV2), 256, 0, stream>>>(k4, md2, Bn, 24, 24, 24);
  count_mask<<<eb(NV2), 256, 0, stream>>>(md2, NV2, cnts + 5);

  // ---- decoder level 4 -> 2 ----
  zero_f32<<<1, 256, 0, stream>>>(st, 256);
  up2_wmma<64, 32, 24, 24, 24><<<cg(NV4, 32, 2), 64, 0, stream>>>(actA, md2, wD42a, ybuf, st);
  bn_elu<32><<<eb((long long)NV2 * 32), 256, 0, stream>>>(ybuf, st, cnts + 5, Pf[28], Pf[29], md2, actB, NV2);
  zero_f32<<<1, 256, 0, stream>>>(st, 256);
  conv3_wmma<32, 32, 48, 48, 48><<<cg(NV2, 32, 4), 128, 0, stream>>>(actB, md2, wD42b, ybuf, st);
  bn_elu<32><<<eb((long long)NV2 * 32), 256, 0, stream>>>(ybuf, st, cnts + 5, Pf[31], Pf[32], md2, actA, NV2);
  cls_prune<32><<<eb(NV2), 256, 0, stream>>>(actA, md2, Pf[41], Pf[42], out + OFF_C2, k2, NV2);
  t_mask<<<eb(NV2), 256, 0, stream>>>(md2, tgt, out + OFF_T2, Bn, 48, 48, 48, 2);
  up_mask_k<<<eb(NV1), 256, 0, stream>>>(k2, md1, Bn, 48, 48, 48);
  count_mask<<<eb(NV1), 256, 0, stream>>>(md1, NV1, cnts + 6);

  // ---- decoder level 2 -> 1 ----
  zero_f32<<<1, 256, 0, stream>>>(st, 256);
  up2_wmma<32, 16, 48, 48, 48><<<cg(NV2, 16, 2), 64, 0, stream>>>(actA, md1, wD21a, ybuf, st);
  bn_elu<16><<<eb((long long)NV1 * 16), 256, 0, stream>>>(ybuf, st, cnts + 6, Pf[34], Pf[35], md1, actB, NV1);
  zero_f32<<<1, 256, 0, stream>>>(st, 256);
  conv3_wmma<16, 16, 96, 96, 96><<<cg(NV1, 16, 4), 128, 0, stream>>>(actB, md1, wD21b, ybuf, st);
  bn_elu<16><<<eb((long long)NV1 * 16), 256, 0, stream>>>(ybuf, st, cnts + 6, Pf[37], Pf[38], md1, actA, NV1);
  cls_prune<16><<<eb(NV1), 256, 0, stream>>>(actA, md1, Pf[43], Pf[44], out + OFF_C1, k1, NV1);
  t_mask<<<eb(NV1), 256, 0, stream>>>(md1, tgt, out + OFF_T1, Bn, 96, 96, 96, 1);

  // ---- final outputs: pruned d1 (f32) and k1 ----
  write_bf_to_f32<<<eb((long long)NV1 * 16), 256, 0, stream>>>(actA, out + OFF_D1, (long long)NV1 * 16);
  write_mask_f32<<<eb(NV1), 256, 0, stream>>>(k1, out + OFF_K1, NV1);
}
